// CausalSelfAttention_38474317037917
// MI455X (gfx1250) — matrix operand order repaired and checked by reading, not compiled, not run on hardware
//
#include <hip/hip_runtime.h>
#include <cstdint>
#include <cstddef>

// Problem constants (match reference)
#define BB 2
#define NN 2048
#define DD 1024
#define HH 16
#define DKV 64
#define HD 1024  // H * DK

typedef __bf16 bf16;
typedef __attribute__((ext_vector_type(16))) __bf16 v16bf;
typedef __attribute__((ext_vector_type(8)))  float  v8f;

union Frag16 { v16bf v; uint4 q[2]; };
union Acc8   { v8f  v; float f[8]; };

// ---------------------------------------------------------------------------
// WMMA helpers (CDNA5 wave32, 16x16x32 bf16 -> f32)
// A layout: lane m=lane&15 is row; lh=lane>>4 selects K-block:
//   v0..3 = K[lh*8 .. lh*8+7], v4..7 = K[16+lh*8 .. 16+lh*8+7]
// B layout: lane n=lane&15 is col; same K split as A (v0..3 = K[lh*8..], v4..7 = K[16+lh*8..])
// C layout: lane n=lane&15 is col; VGPR v holds row (v + 8*lh)
// ---------------------------------------------------------------------------
__device__ __forceinline__ v16bf load_a_frag(const bf16* __restrict__ base, int ld,
                                             int row0, int k0, int lane) {
  const int m = lane & 15, lh = lane >> 4;
  const bf16* p = base + (size_t)(row0 + m) * ld + k0;
  Frag16 f;
  f.q[0] = *(const uint4*)(p + lh * 8);
  f.q[1] = *(const uint4*)(p + 16 + lh * 8);
  return f.v;
}

__device__ __forceinline__ v16bf load_b_frag(const bf16* __restrict__ base, int ld,
                                             int n0, int k0, int lane) {
  const int n = lane & 15, lh = lane >> 4;
  const bf16* p = base + (size_t)(n0 + n) * ld + k0;
  Frag16 f;
  f.q[0] = *(const uint4*)(p + lh * 8);
  f.q[1] = *(const uint4*)(p + 16 + lh * 8);
  return f.v;
}

__device__ __forceinline__ v8f wmma_bf(v16bf a, v16bf b, v8f c) {
  return __builtin_amdgcn_wmma_f32_16x16x32_bf16(
      /*neg_a=*/false, a, /*neg_b=*/false, b,
      /*c_mod=*/(short)0, c, /*reuse_a=*/false, /*reuse_b=*/false);
}

__device__ __forceinline__ float redmax16(float x) {
  #pragma unroll
  for (int m = 1; m < 16; m <<= 1) x = fmaxf(x, __shfl_xor(x, m, 32));
  return x;
}
__device__ __forceinline__ float redsum16(float x) {
  #pragma unroll
  for (int m = 1; m < 16; m <<= 1) x += __shfl_xor(x, m, 32);
  return x;
}

// ---------------------------------------------------------------------------
// CDNA5 async global->LDS copy (ASYNCcnt-tracked, no VGPR data path).
// Per-lane: LDS[lds_off] = MEM[gaddr], 16 bytes.
// ---------------------------------------------------------------------------
__device__ __forceinline__ void async_copy_b128(uint32_t lds_off, uint64_t gaddr) {
  asm volatile("global_load_async_to_lds_b128 %0, %1, off"
               :: "v"(lds_off), "v"(gaddr) : "memory");
}
__device__ __forceinline__ void wait_async0() {
  asm volatile("s_wait_asynccnt 0" ::: "memory");
}

// Cooperative stage of one B tile [64 cols][32 k] bf16 (4KB) into LDS.
// 128 threads x 2 chunks x 16B. ldsbuf laid out [col][k], row stride 32 elems.
__device__ __forceinline__ void stage_b_async(const bf16* __restrict__ WT, int ldw,
                                              int colbase, int k0,
                                              bf16* ldsbuf, int tid) {
  #pragma unroll
  for (int i = 0; i < 2; ++i) {
    const int ch  = tid * 2 + i;       // 0..255
    const int col = ch >> 2;           // 0..63
    const int ko  = (ch & 3) * 8;      // 0,8,16,24 (bf16 elems -> 16B chunks)
    const bf16* g = WT + (size_t)(colbase + col) * ldw + k0 + ko;
    async_copy_b128((uint32_t)(uintptr_t)(ldsbuf + col * 32 + ko),
                    (uint64_t)(uintptr_t)g);
  }
}

// ---------------------------------------------------------------------------
// Conversion kernels
// ---------------------------------------------------------------------------
__global__ void cvt_f32_to_bf16(const float* __restrict__ src, bf16* __restrict__ dst, int n) {
  int i = blockIdx.x * blockDim.x + threadIdx.x;
  const int stride = gridDim.x * blockDim.x;
  for (; i < n; i += stride) dst[i] = (bf16)src[i];
}

// W [rows, cols] row-major (f32) -> WT [cols, rows] row-major (bf16)
__global__ void cvt_transpose_bf16(const float* __restrict__ W, bf16* __restrict__ WT,
                                   int rows, int cols) {
  const int c = blockIdx.x * blockDim.x + threadIdx.x;
  const int r = blockIdx.y;
  if (c < cols) WT[(size_t)c * rows + r] = (bf16)W[(size_t)r * cols + c];
}

// ---------------------------------------------------------------------------
// Fused QKV projection: Xb[B*N, D] @ WT -> Q/K [b,h,n,dk] bf16 and V^T [b,h,dv,n]
// Block = 128 threads (4 waves); block tile = 128 rows x 64 cols; each wave
// does 32 rows. B tile (shared by all 4 waves) is double-buffered in LDS via
// async global->LDS copies.
// grid: (16 col-strips, B*N/128, 3 which)
// ---------------------------------------------------------------------------
__global__ __launch_bounds__(128) void gemm_qkv_kernel(
    const bf16* __restrict__ Xb,
    const bf16* __restrict__ WqT, const bf16* __restrict__ WkT, const bf16* __restrict__ WvT,
    const float* __restrict__ bq, const float* __restrict__ bk, const float* __restrict__ bv,
    bf16* __restrict__ Qb, bf16* __restrict__ Kb, bf16* __restrict__ Vt) {
  __shared__ __align__(16) bf16 Bt[2][64][32];
  const int tid = threadIdx.x, wave = tid >> 5, lane = tid & 31;
  const int lh = lane >> 4, ln = lane & 15;
  const int which = blockIdx.z;
  const bf16*  WT   = (which == 0) ? WqT : (which == 1) ? WkT : WvT;
  const float* bias = (which == 0) ? bq  : (which == 1) ? bk  : bv;
  const int colbase = blockIdx.x * 64;
  const int rowbase = (blockIdx.y * 4 + wave) * 32;  // in [0, B*N)

  Acc8 acc[2][4];
  #pragma unroll
  for (int rr = 0; rr < 2; ++rr)
    #pragma unroll
    for (int t = 0; t < 4; ++t)
      #pragma unroll
      for (int i = 0; i < 8; ++i) acc[rr][t].f[i] = 0.f;

  stage_b_async(WT, DD, colbase, 0, &Bt[0][0][0], tid);
  wait_async0();
  __syncthreads();

  int cur = 0;
  for (int k0 = 0; k0 < DD; k0 += 32) {
    if (k0 + 32 < DD)
      stage_b_async(WT, DD, colbase, k0 + 32, &Bt[cur ^ 1][0][0], tid);
    const v16bf a0 = load_a_frag(Xb, DD, rowbase,      k0, lane);
    const v16bf a1 = load_a_frag(Xb, DD, rowbase + 16, k0, lane);
    #pragma unroll
    for (int t = 0; t < 4; ++t) {
      const v16bf bfr = load_b_frag(&Bt[cur][0][0], 32, t * 16, 0, lane);
      acc[0][t].v = wmma_bf(a0, bfr, acc[0][t].v);
      acc[1][t].v = wmma_bf(a1, bfr, acc[1][t].v);
    }
    wait_async0();
    __syncthreads();
    cur ^= 1;
  }

  const int b    = rowbase / NN;
  const int rloc = rowbase % NN;
  const int h    = blockIdx.x;  // 64-wide col strip == one head
  #pragma unroll
  for (int rr = 0; rr < 2; ++rr) {
    #pragma unroll
    for (int t = 0; t < 4; ++t) {
      const int col = colbase + t * 16 + ln;
      const int dk  = col & 63;
      const float bs = bias[col];
      #pragma unroll
      for (int v = 0; v < 8; ++v) {
        const int r = rloc + rr * 16 + v + 8 * lh;
        const float val = acc[rr][t].f[v] + bs;
        if (which == 0)
          Qb[((size_t)(b * HH + h) * NN + r) * DKV + dk] = (bf16)val;
        else if (which == 1)
          Kb[((size_t)(b * HH + h) * NN + r) * DKV + dk] = (bf16)val;
        else
          Vt[((size_t)(b * HH + h) * DKV + dk) * NN + r] = (bf16)val;
      }
    }
  }
}

// ---------------------------------------------------------------------------
// Flash-attention: each wave owns one 16-row Q tile of one (b,h).
// grid: (N/64, B*H), block 128 (4 waves). Online softmax; LDS transposes the
// probability tile from C-layout to A-layout for the P·V WMMAs.
// ---------------------------------------------------------------------------
__global__ __launch_bounds__(128) void attn_kernel(
    const bf16* __restrict__ Qb, const bf16* __restrict__ Kb,
    const bf16* __restrict__ Vt, bf16* __restrict__ AVb) {
  __shared__ __align__(16) bf16 pbuf[4][16][32];
  const int tid = threadIdx.x, wave = tid >> 5, lane = tid & 31;
  const int lh = lane >> 4, ln = lane & 15;
  const int bh = blockIdx.y;
  const int b = bh / HH, h = bh % HH;
  const int rowbase = (blockIdx.x * 4 + wave) * 16;

  const bf16* Qh = Qb + (size_t)bh * NN * DKV;
  const bf16* Kh = Kb + (size_t)bh * NN * DKV;
  const bf16* Vh = Vt + (size_t)bh * DKV * NN;

  const v16bf qf0 = load_a_frag(Qh, DKV, rowbase, 0, lane);
  const v16bf qf1 = load_a_frag(Qh, DKV, rowbase, 32, lane);

  float mrow[8], lrow[8], alpha[8];
  Acc8 o[4];
  #pragma unroll
  for (int v = 0; v < 8; ++v) { mrow[v] = -__builtin_inff(); lrow[v] = 0.f; }
  #pragma unroll
  for (int t = 0; t < 4; ++t)
    #pragma unroll
    for (int i = 0; i < 8; ++i) o[t].f[i] = 0.f;

  const int rmax = rowbase + 15;
  for (int j0 = 0; j0 <= rmax; j0 += 32) {
    Acc8 s0, s1;
    #pragma unroll
    for (int i = 0; i < 8; ++i) { s0.f[i] = 0.f; s1.f[i] = 0.f; }
    {
      const v16bf kb00 = load_b_frag(Kh, DKV, j0,      0,  lane);
      const v16bf kb01 = load_b_frag(Kh, DKV, j0,      32, lane);
      s0.v = wmma_bf(qf0, kb00, s0.v);
      s0.v = wmma_bf(qf1, kb01, s0.v);
      const v16bf kb10 = load_b_frag(Kh, DKV, j0 + 16, 0,  lane);
      const v16bf kb11 = load_b_frag(Kh, DKV, j0 + 16, 32, lane);
      s1.v = wmma_bf(qf0, kb10, s1.v);
      s1.v = wmma_bf(qf1, kb11, s1.v);
    }
    #pragma unroll
    for (int v = 0; v < 8; ++v) {
      const int r = rowbase + v + 8 * lh;
      float x0 = s0.f[v] * 0.125f;                 // 1/sqrt(64)
      float x1 = s1.f[v] * 0.125f;
      if (j0 + ln      > r) x0 = -__builtin_inff();  // causal mask
      if (j0 + 16 + ln > r) x1 = -__builtin_inff();
      const float mt   = redmax16(fmaxf(x0, x1));
      const float mnew = fmaxf(mrow[v], mt);
      alpha[v] = __expf(mrow[v] - mnew);
      const float p0 = __expf(x0 - mnew);
      const float p1 = __expf(x1 - mnew);
      pbuf[wave][v + 8 * lh][ln]      = (bf16)p0;
      pbuf[wave][v + 8 * lh][16 + ln] = (bf16)p1;
      const float rs = redsum16(p0 + p1);
      lrow[v] = lrow[v] * alpha[v] + rs;
      mrow[v] = mnew;
    }
    __builtin_amdgcn_wave_barrier();   // LDS in-order within wave; block reordering
    const v16bf pf = load_a_frag(&pbuf[wave][0][0], 32, 0, 0, lane);
    __builtin_amdgcn_wave_barrier();
    #pragma unroll
    for (int t = 0; t < 4; ++t) {
      const v16bf vf = load_b_frag(Vh, NN, t * 16, j0, lane);
      #pragma unroll
      for (int v = 0; v < 8; ++v) o[t].f[v] *= alpha[v];
      o[t].v = wmma_bf(pf, vf, o[t].v);
    }
  }

  float inv[8];
  #pragma unroll
  for (int v = 0; v < 8; ++v) inv[v] = 1.0f / lrow[v];
  #pragma unroll
  for (int t = 0; t < 4; ++t) {
    const int dv = t * 16 + ln;
    #pragma unroll
    for (int v = 0; v < 8; ++v) {
      const int r = rowbase + v + 8 * lh;
      AVb[((size_t)(b * NN + r)) * HD + h * DKV + dv] = (bf16)(o[t].f[v] * inv[v]);
    }
  }
}

// ---------------------------------------------------------------------------
// Output projection: AVb[B*N, HD] @ WoT -> out f32 [B*N, D] (+ bo)
// Same LDS-staged, async double-buffered structure as the QKV GEMM.
// ---------------------------------------------------------------------------
__global__ __launch_bounds__(128) void gemm_out_kernel(
    const bf16* __restrict__ AVb, const bf16* __restrict__ WoT,
    const float* __restrict__ bo, float* __restrict__ out) {
  __shared__ __align__(16) bf16 Bt[2][64][32];
  const int tid = threadIdx.x, wave = tid >> 5, lane = tid & 31;
  const int lh = lane >> 4, ln = lane & 15;
  const int colbase = blockIdx.x * 64;
  const int rowbase = (blockIdx.y * 4 + wave) * 32;

  Acc8 acc[2][4];
  #pragma unroll
  for (int rr = 0; rr < 2; ++rr)
    #pragma unroll
    for (int t = 0; t < 4; ++t)
      #pragma unroll
      for (int i = 0; i < 8; ++i) acc[rr][t].f[i] = 0.f;

  stage_b_async(WoT, HD, colbase, 0, &Bt[0][0][0], tid);
  wait_async0();
  __syncthreads();

  int cur = 0;
  for (int k0 = 0; k0 < HD; k0 += 32) {
    if (k0 + 32 < HD)
      stage_b_async(WoT, HD, colbase, k0 + 32, &Bt[cur ^ 1][0][0], tid);
    const v16bf a0 = load_a_frag(AVb, HD, rowbase,      k0, lane);
    const v16bf a1 = load_a_frag(AVb, HD, rowbase + 16, k0, lane);
    #pragma unroll
    for (int t = 0; t < 4; ++t) {
      const v16bf bfr = load_b_frag(&Bt[cur][0][0], 32, t * 16, 0, lane);
      acc[0][t].v = wmma_bf(a0, bfr, acc[0][t].v);
      acc[1][t].v = wmma_bf(a1, bfr, acc[1][t].v);
    }
    wait_async0();
    __syncthreads();
    cur ^= 1;
  }

  #pragma unroll
  for (int rr = 0; rr < 2; ++rr) {
    #pragma unroll
    for (int t = 0; t < 4; ++t) {
      const int col = colbase + t * 16 + ln;
      const float bs = bo[col];
      #pragma unroll
      for (int v = 0; v < 8; ++v) {
        const int rg = rowbase + rr * 16 + v + 8 * lh;  // global row over B*N
        out[(size_t)rg * DD + col] = acc[rr][t].f[v] + bs;
      }
    }
  }
}

// ---------------------------------------------------------------------------
extern "C" void kernel_launch(void* const* d_in, const int* in_sizes, int n_in,
                              void* d_out, int out_size, void* d_ws, size_t ws_size,
                              hipStream_t stream) {
  const float* X  = (const float*)d_in[0];
  const float* Wq = (const float*)d_in[1];
  const float* bq = (const float*)d_in[2];
  const float* Wk = (const float*)d_in[3];
  const float* bk = (const float*)d_in[4];
  const float* Wv = (const float*)d_in[5];
  const float* bv = (const float*)d_in[6];
  const float* Wo = (const float*)d_in[7];
  const float* bo = (const float*)d_in[8];
  float* out = (float*)d_out;

  char* ws = (char*)d_ws;
  size_t off = 0;
  auto take = [&](size_t bytes) -> char* {
    char* p = ws + off;
    off += (bytes + 255) & ~(size_t)255;
    return p;
  };
  bf16* Xb  = (bf16*)take((size_t)BB * NN * DD * 2);   // X in bf16
  bf16* WqT = (bf16*)take((size_t)DD * HD * 2);
  bf16* WkT = (bf16*)take((size_t)DD * HD * 2);
  bf16* WvT = (bf16*)take((size_t)DD * HD * 2);
  bf16* WoT = (bf16*)take((size_t)HD * DD * 2);
  bf16* Qb  = (bf16*)take((size_t)BB * HH * NN * DKV * 2);
  bf16* Kb  = (bf16*)take((size_t)BB * HH * NN * DKV * 2);
  bf16* Vt  = (bf16*)take((size_t)BB * HH * DKV * NN * 2);
  bf16* AVb = (bf16*)take((size_t)BB * NN * HD * 2);
  (void)ws_size; (void)in_sizes; (void)n_in; (void)out_size;

  // 1) fp32 -> bf16 conversions (X) and weight transposes (so GEMM B-frags are
  //    contiguous 16B loads along K)
  cvt_f32_to_bf16<<<2048, 256, 0, stream>>>(X, Xb, BB * NN * DD);
  {
    dim3 blk(256);
    cvt_transpose_bf16<<<dim3(HD / 256, DD), blk, 0, stream>>>(Wq, WqT, DD, HD);
    cvt_transpose_bf16<<<dim3(HD / 256, DD), blk, 0, stream>>>(Wk, WkT, DD, HD);
    cvt_transpose_bf16<<<dim3(HD / 256, DD), blk, 0, stream>>>(Wv, WvT, DD, HD);
    cvt_transpose_bf16<<<dim3(DD / 256, HD), blk, 0, stream>>>(Wo, WoT, HD, DD);
  }

  // 2) fused QKV projection (z = which of Q/K/V), async-LDS staged B
  gemm_qkv_kernel<<<dim3(HD / 64, (BB * NN) / 128, 3), 128, 0, stream>>>(
      Xb, WqT, WkT, WvT, bq, bk, bv, Qb, Kb, Vt);

  // 3) causal flash attention per (b,h)
  attn_kernel<<<dim3(NN / 64, BB * HH), 128, 0, stream>>>(Qb, Kb, Vt, AVb);

  // 4) output projection + bias -> f32 d_out
  gemm_out_kernel<<<dim3(DD / 64, (BB * NN) / 128), 128, 0, stream>>>(AVb, WoT, bo, out);
}
